// ExpertsMLP_1314259992886
// MI455X (gfx1250) — compile-verified, matrix operand order checked by reading, not run on hardware
//
#include <hip/hip_runtime.h>
#include <math.h>

typedef __attribute__((ext_vector_type(16))) __bf16 v16bf;
typedef __attribute__((ext_vector_type(8)))  float  v8f;

constexpr int E = 32, C = 4096, D = 256, H = 1024;
constexpr int BM = 32;                 // rows of C per block
constexpr int SLAB = 256;              // H columns per slab
constexpr int NSLAB = H / SLAB;        // 4
constexpr int KSTEP = 32;              // K per WMMA
constexpr int BLOCKS_PER_E = C / BM;   // 128
constexpr int NBLOCKS = E * BLOCKS_PER_E; // 4096
constexpr int HS_STRIDE = 264;         // 256 + 8 pad (bf16 elems)
constexpr int WP_STRIDE = 40;          // 32 + 8 pad (bf16 elems); 80B row, banks spread
constexpr int PANEL_BYTES = SLAB * WP_STRIDE * 2; // 20480
constexpr size_t NY = (size_t)E * C * D;
constexpr size_t WS_W1T_OFF = 1 << 16;            // partials live in [0, 64KB)
constexpr size_t WELEMS = (size_t)E * D * H;      // 8.4M bf16 elems per weight

// fp32 -> bf16 round-to-nearest-even
__device__ __forceinline__ unsigned short f2bf(float f) {
  unsigned u = __float_as_uint(f);
  u += 0x7FFFu + ((u >> 16) & 1u);
  return (unsigned short)(u >> 16);
}
__device__ __forceinline__ unsigned pack2bf(float a, float b) {
  return (unsigned)f2bf(a) | ((unsigned)f2bf(b) << 16);
}

union Frag { unsigned u[8]; v16bf v; };

__device__ __forceinline__ v8f zero8() {
  v8f z;
#pragma unroll
  for (int i = 0; i < 8; ++i) z[i] = 0.0f;
  return z;
}

// Wait for this wave's async LDS copies + LDS ops, then full workgroup barrier.
// Returns a zero token (in an SGPR) that callers fold into LDS addresses so the
// scheduler cannot hoist dependent ds_loads above the wait/barrier.
__device__ __forceinline__ unsigned wait_async_and_barrier() {
  unsigned dep;
  asm volatile(
      "s_wait_asynccnt 0x0\n\t"
      "s_wait_dscnt 0x0\n\t"
      "s_barrier_signal -1\n\t"
      "s_barrier_wait -1\n\t"
      "s_mov_b32 %0, 0"
      : "=s"(dep)
      :
      : "memory");
  return dep;
}

// ---------------- prologue: fp32 [E][R][Cc] -> bf16 transposed [E][Cc][R] ----
__global__ __launch_bounds__(256) void transpose_bf16_kernel(
    const float* __restrict__ src, unsigned short* __restrict__ dst,
    int R, int Cc)
{
  __shared__ unsigned short tile[32][33];
  const int tilesR = R >> 5, tilesC = Cc >> 5;
  int b = blockIdx.x;
  const int e  = b / (tilesR * tilesC);
  int rem      = b % (tilesR * tilesC);
  const int r0 = (rem / tilesC) * 32;
  const int c0 = (rem % tilesC) * 32;
  const int t  = threadIdx.x;
#pragma unroll
  for (int j = 0; j < 4; ++j) {
    int idx = t + 256 * j;
    int rr = idx >> 5, cc = idx & 31;
    tile[rr][cc] = f2bf(src[((size_t)e * R + r0 + rr) * Cc + c0 + cc]);
  }
  __syncthreads();
#pragma unroll
  for (int j = 0; j < 4; ++j) {
    int idx = t + 256 * j;
    int rr = idx >> 5, cc = idx & 31;
    dst[((size_t)e * Cc + c0 + rr) * R + r0 + cc] = tile[cc][rr];
  }
}

// ---------------- main fused MLP kernel ----------------
__global__ __launch_bounds__(256) void experts_mlp_kernel(
    const float* __restrict__ x,
    const unsigned short* __restrict__ w1t,   // bf16 [E][H][D]
    const float* __restrict__ b1,
    const unsigned short* __restrict__ w2t,   // bf16 [E][D][H]
    const float* __restrict__ b2,
    const float* __restrict__ bungee,
    float* __restrict__ out, float* __restrict__ partials)
{
  __shared__ unsigned short Hs[BM * HS_STRIDE];                     // 16896 B
  __shared__ __align__(16) unsigned short Wp[2 * SLAB * WP_STRIDE]; // 40960 B, dbl-buffered

  const int tid   = threadIdx.x;
  const int lane  = tid & 31;
  const int wid   = tid >> 5;
  const int lhalf = lane >> 4;
  const int l15   = lane & 15;
  const int bid   = blockIdx.x;
  const int e     = bid / BLOCKS_PER_E;
  const int c0    = (bid % BLOCKS_PER_E) * BM;

  const int m0    = (wid & 1) * 16;
  const int nbase = (wid >> 1) * 64;

  // LDS byte offset of Wp for async destinations (low 32 bits of flat LDS addr)
  const unsigned wp_base = (unsigned)(unsigned long long)&Wp[0];

  // ---- preload GEMM1 A-fragments for this wave's 16 rows (reused all slabs) ----
  Frag afragX[8];
  {
    const float* xr = x + ((size_t)e * C + c0 + m0 + l15) * D;
#pragma unroll
    for (int kk = 0; kk < 8; ++kk) {
#pragma unroll
      for (int v = 0; v < 8; ++v) {
        int kb = (v >> 2) * 16 + lhalf * 8 + (v & 3) * 2;
        float2 tv = *(const float2*)(xr + kk * KSTEP + kb);
        afragX[kk].u[v] = pack2bf(tv.x, tv.y);
      }
    }
  }

  const float bg = bungee[e];

  v8f yacc[4];
#pragma unroll
  for (int t = 0; t < 4; ++t) yacc[t] = zero8();

  // async panel issue: 256 threads x 1 row (64B) x 4 chunks of 16B
  auto issue_panel = [&](const unsigned short* gbase, size_t row_stride_elems, int buf) {
    const unsigned short* g = gbase + (size_t)tid * row_stride_elems;
    unsigned lds = wp_base + (unsigned)(buf * PANEL_BYTES) + (unsigned)(tid * (WP_STRIDE * 2));
#pragma unroll
    for (int j = 0; j < 4; ++j) {
      asm volatile("global_load_async_to_lds_b128 %0, %1, off"
                   :: "v"(lds + j * 16), "v"(g + j * 8) : "memory");
    }
  };

  for (int s = 0; s < NSLAB; ++s) {
    const int hbase = s * SLAB;
    v8f acc[4];
#pragma unroll
    for (int t = 0; t < 4; ++t) acc[t] = zero8();

    // ======== GEMM1: h_slab = X @ W1[:, hbase:hbase+256] ========
    // panel kk: rows n=h (256), k=d chunk; w1t[e][hbase+n][kk*32 + k]
    issue_panel(w1t + ((size_t)e * H + hbase) * D + 0 * KSTEP, D, 0);
#pragma unroll
    for (int kk = 0; kk < 8; ++kk) {
      const int dep  = (int)wait_async_and_barrier();     // panel kk visible to all waves
      if (kk < 7)
        issue_panel(w1t + ((size_t)e * H + hbase) * D + (kk + 1) * KSTEP, D, (kk + 1) & 1);
      const int bufo = (kk & 1) * SLAB * WP_STRIDE + dep; // dep==0; forces load-after-barrier
#pragma unroll
      for (int t = 0; t < 4; ++t) {
        Frag bfr;
        int n0 = nbase + t * 16;
#pragma unroll
        for (int v = 0; v < 8; ++v) {
          int kb = (v >> 2) * 16 + lhalf * 8 + (v & 3) * 2;
          bfr.u[v] = *(const unsigned*)&Wp[bufo + (n0 + l15) * WP_STRIDE + kb];
        }
        acc[t] = __builtin_amdgcn_wmma_f32_16x16x32_bf16(
            false, afragX[kk].v, false, bfr.v, (short)0, acc[t], false, false);
      }
    }

    // bias1 + exact GELU -> Hs (bf16)
#pragma unroll
    for (int t = 0; t < 4; ++t) {
      int n0 = nbase + t * 16;
      float b1v = b1[(size_t)e * H + hbase + n0 + l15];
#pragma unroll
      for (int v = 0; v < 8; ++v) {
        int m = v + 8 * lhalf;
        float val = acc[t][v] + b1v;
        float g = 0.5f * val * (1.0f + erff(val * 0.70710678118654752f));
        Hs[(m0 + m) * HS_STRIDE + n0 + l15] = f2bf(g);
      }
    }

    // ======== GEMM2: Y += gelu(h_slab) @ W2[hbase:hbase+256, :] ========
    // panel kk: rows n=d (256), k=h chunk; w2t[e][n][hbase + kk*32 + k]
    issue_panel(w2t + (size_t)e * D * H + hbase + 0 * KSTEP, H, 0);
    for (int kk = 0; kk < 8; ++kk) {
      const int dep  = (int)wait_async_and_barrier();  // panel kk + all Hs writes visible
      if (kk < 7)
        issue_panel(w2t + (size_t)e * D * H + hbase + (kk + 1) * KSTEP, H, (kk + 1) & 1);
      const int bufo = (kk & 1) * SLAB * WP_STRIDE + dep;
      Frag afr;
#pragma unroll
      for (int v = 0; v < 8; ++v) {
        int kb = (v >> 2) * 16 + lhalf * 8 + (v & 3) * 2;
        afr.u[v] = *(const unsigned*)&Hs[dep + (m0 + l15) * HS_STRIDE + kk * KSTEP + kb];
      }
#pragma unroll
      for (int t = 0; t < 4; ++t) {
        Frag bfr;
        int n0 = nbase + t * 16;
#pragma unroll
        for (int v = 0; v < 8; ++v) {
          int kb = (v >> 2) * 16 + lhalf * 8 + (v & 3) * 2;
          bfr.u[v] = *(const unsigned*)&Wp[bufo + (n0 + l15) * WP_STRIDE + kb];
        }
        yacc[t] = __builtin_amdgcn_wmma_f32_16x16x32_bf16(
            false, afr.v, false, bfr.v, (short)0, yacc[t], false, false);
      }
    }
  }

  // ---- epilogue: bias2, bungee scale, store y, sum-of-squares ----
  float ssq = 0.0f;
#pragma unroll
  for (int t = 0; t < 4; ++t) {
    int n0 = nbase + t * 16;
    float b2v = b2[(size_t)e * D + n0 + l15];
#pragma unroll
    for (int v = 0; v < 8; ++v) {
      int m = v + 8 * lhalf;
      float val = (yacc[t][v] + b2v) * bg;
      out[((size_t)e * C + c0 + m0 + m) * D + n0 + l15] = val;
      ssq += val * val;
    }
  }
  __syncthreads();                 // Wp no longer needed; reuse as reduction buffer
  float* red = (float*)&Wp[0];
  red[tid] = ssq;
  __syncthreads();
#pragma unroll
  for (int sft = 128; sft > 0; sft >>= 1) {
    if (tid < sft) red[tid] += red[tid + sft];
    __syncthreads();
  }
  if (tid == 0) partials[bid] = red[0];
}

__global__ __launch_bounds__(256) void finalize_kernel(
    const float* __restrict__ partials, const float* __restrict__ bungee,
    float* __restrict__ out)
{
  __shared__ float red[256];
  int tid = threadIdx.x;
  float s = 0.0f;
  for (int i = tid; i < NBLOCKS; i += 256) s += partials[i];
  red[tid] = s;
  __syncthreads();
#pragma unroll
  for (int sft = 128; sft > 0; sft >>= 1) {
    if (tid < sft) red[tid] += red[tid + sft];
    __syncthreads();
  }
  if (tid == 0) {
    float rms = sqrtf(red[0] / (float)NY);
    float mn = bungee[0], mx = bungee[0], sm = 0.0f;
    for (int i = 0; i < E; ++i) {
      float v = bungee[i];
      sm += v;
      mn = fminf(mn, v);
      mx = fmaxf(mx, v);
    }
    out[NY + 0] = rms;
    out[NY + 1] = sm / (float)E;
    out[NY + 2] = mn;
    out[NY + 3] = mx;
  }
}

extern "C" void kernel_launch(void* const* d_in, const int* in_sizes, int n_in,
                              void* d_out, int out_size, void* d_ws, size_t ws_size,
                              hipStream_t stream) {
  const float* x      = (const float*)d_in[0];
  const float* w1     = (const float*)d_in[1];
  const float* b1     = (const float*)d_in[2];
  const float* w2     = (const float*)d_in[3];
  const float* b2     = (const float*)d_in[4];
  const float* bungee = (const float*)d_in[5];
  float* out = (float*)d_out;

  float* partials      = (float*)d_ws;                                // 4096 floats
  unsigned short* w1t  = (unsigned short*)((char*)d_ws + WS_W1T_OFF); // bf16 [E][H][D]
  unsigned short* w2t  = w1t + WELEMS;                                // bf16 [E][D][H]

  // prologue: convert + transpose weights to bf16 panel-friendly layouts
  hipLaunchKernelGGL(transpose_bf16_kernel, dim3(E * (D / 32) * (H / 32)), dim3(256),
                     0, stream, w1, w1t, D, H);
  hipLaunchKernelGGL(transpose_bf16_kernel, dim3(E * (H / 32) * (D / 32)), dim3(256),
                     0, stream, w2, w2t, H, D);

  hipLaunchKernelGGL(experts_mlp_kernel, dim3(NBLOCKS), dim3(256), 0, stream,
                     x, w1t, b1, w2t, b2, bungee, out, partials);
  hipLaunchKernelGGL(finalize_kernel, dim3(1), dim3(256), 0, stream,
                     partials, bungee, out);
}
